// LSTMAggregator_23510650978751
// MI455X (gfx1250) — matrix operand order repaired and checked by reading, not compiled, not run on hardware
//
#include <hip/hip_runtime.h>
#include <stdint.h>

// ---------------- types ----------------
typedef __attribute__((ext_vector_type(16))) __bf16        v16bf;
typedef __attribute__((ext_vector_type(8)))  float         v8f;
typedef __attribute__((ext_vector_type(4)))  float         f32x4;
typedef __attribute__((ext_vector_type(4)))  unsigned int  u32x4;
typedef __attribute__((ext_vector_type(2)))  unsigned int  u32x2;

// ---------------- problem constants ----------------
#define N_SEQ    8192
#define T_SEQ    32
#define D_IN     256
#define H_DIM    256
#define G_DIM    1024          // 4*H
#define K_TOT    512           // D + H
#define M_TILE   64            // sequences per block
#define NTHREADS 512           // 16 waves
#define A_STRIDE 1040          // bytes per LDS A row: 512 bf16 (1024B) + 16B pad (bank-conflict free)

// LDS layout offsets (bytes)
#define OFF_A    0
#define OFF_H32  (M_TILE * A_STRIDE)                 // 66560
#define OFF_CS   (OFF_H32 + M_TILE * H_DIM * 4)     // +65536 = 132096
#define OFF_LEN  (OFF_CS  + M_TILE * H_DIM * 4)     // +65536 = 197632
#define LDS_SIZE (OFF_LEN + M_TILE * 4)             // 197888

__device__ __forceinline__ unsigned short f2bf(float f) {
  union { float f; uint32_t u; } v; v.f = f;
  uint32_t r = v.u + 0x7FFFu + ((v.u >> 16) & 1u);   // round-to-nearest-even
  return (unsigned short)(r >> 16);
}
__device__ __forceinline__ float sigmf(float x) { return 1.0f / (1.0f + __expf(-x)); }
__device__ __forceinline__ float tanh_fast(float x) { return 2.0f * sigmf(2.0f * x) - 1.0f; }

// ---------------- weight packing: W = [W_ih | W_hh] -> bf16, bias = b_ih + b_hh ----------------
__global__ void lstm_pack_weights(const float* __restrict__ Wih, const float* __restrict__ Whh,
                                  const float* __restrict__ bih, const float* __restrict__ bhh,
                                  unsigned short* __restrict__ wpk, float* __restrict__ bias) {
  int i = blockIdx.x * blockDim.x + threadIdx.x;
  if (i < G_DIM * K_TOT) {
    int n = i >> 9;            // gate row 0..1023
    int k = i & (K_TOT - 1);   // 0..511
    float v = (k < D_IN) ? Wih[n * D_IN + k] : Whh[n * H_DIM + (k - D_IN)];
    wpk[i] = f2bf(v);
  }
  if (i < G_DIM) bias[i] = bih[i] + bhh[i];
}

// ---------------- main fused LSTM kernel ----------------
__global__ __launch_bounds__(NTHREADS, 1)
void lstm_wmma_kernel(const float* __restrict__ msg, const int* __restrict__ lengths,
                      const int* __restrict__ tstamps, const unsigned short* __restrict__ wpk,
                      const float* __restrict__ bias, float* __restrict__ out) {
  extern __shared__ __align__(16) char smem[];
  char*  Abuf  = smem + OFF_A;                 // [64 rows][A_STRIDE] bf16 [x | h]
  float* h32   = (float*)(smem + OFF_H32);     // [64][256] f32 h state
  float* csL   = (float*)(smem + OFF_CS);      // [64][256] f32 c state
  int*   slenL = (int*)(smem + OFF_LEN);       // [64] lengths

  const int tid  = (int)threadIdx.x;
  const int wv   = tid >> 5;
  const int lane = tid & 31;
  const int lh   = lane & 15;          // lane 0..15 within half-wave
  const int ls   = lane >> 4;          // which half of the wave
  const int row0 = (int)blockIdx.x * M_TILE;
  const int jcol = wv * 16 + lh;       // this lane's h/gate column (0..255)

  // ---- zero h region of A (bytes 512..1023 per row) ----
  #pragma unroll
  for (int i = 0; i < 4; ++i) {
    int u = tid + i * NTHREADS;        // 2048 x 16-byte units
    int row = u >> 5, c = u & 31;
    *(u32x4*)(Abuf + row * A_STRIDE + 512 + c * 16) = (u32x4){0u, 0u, 0u, 0u};
  }
  // ---- zero c-state ----
  #pragma unroll
  for (int i = 0; i < 8; ++i) {
    int u = tid + i * NTHREADS;        // 4096 x 16-byte units
    ((u32x4*)csL)[u] = (u32x4){0u, 0u, 0u, 0u};
  }
  // ---- lengths into LDS ----
  if (tid < M_TILE) slenL[tid] = lengths[row0 + tid];

  float bv[4];
  #pragma unroll
  for (int g = 0; g < 4; ++g) bv[g] = bias[g * 256 + jcol];

  v8f C[4][4];                          // [gate group][m-tile] accumulators (128 VGPRs)
  const char* wb = (const char*)wpk;
  const int jrow_off = jcol << 10;      // byte offset of this lane's weight row within a group
  __syncthreads();

  for (int t = 0; t < T_SEQ; ++t) {
    // ---- stage x_t (f32 -> bf16) into A cols 0..255 ----
    #pragma unroll
    for (int i = 0; i < 8; ++i) {
      int u = tid + i * NTHREADS;       // 4096 float4 units
      int row = u >> 6, c4 = u & 63;
      f32x4 x4 = *(const f32x4*)(msg + (((size_t)(row0 + row)) * T_SEQ + t) * D_IN + c4 * 4);
      u32x2 p;
      p.x = (uint32_t)f2bf(x4.x) | ((uint32_t)f2bf(x4.y) << 16);
      p.y = (uint32_t)f2bf(x4.z) | ((uint32_t)f2bf(x4.w) << 16);
      *(u32x2*)(Abuf + row * A_STRIDE + c4 * 8) = p;
    }
    __syncthreads();                    // x staged; previous step's h writes visible

    // ---- init accumulators with fused bias ----
    #pragma unroll
    for (int g = 0; g < 4; ++g) {
      v8f cinit;
      #pragma unroll
      for (int r = 0; r < 8; ++r) cinit[r] = bv[g];
      #pragma unroll
      for (int m = 0; m < 4; ++m) C[g][m] = cinit;
    }

    // ---- gates += [x|h] @ W^T  (bf16 WMMA 16x16x32, f32 acc) ----
    // Per K-tile: hold B for all 4 gate groups (32 VGPRs); each A-frag feeds 4 WMMAs.
    #pragma unroll 2
    for (int kg = 0; kg < 16; ++kg) {
      v16bf Bg[4];
      #pragma unroll
      for (int g = 0; g < 4; ++g)
        Bg[g] = *(const v16bf*)(wb + ((g * 256) << 10) + jrow_off + (kg << 6) + (ls << 5));
      #pragma unroll
      for (int m = 0; m < 4; ++m) {
        union { v16bf v; u32x4 q[2]; } a;
        const char* ab = Abuf + (m * 16 + lh) * A_STRIDE + (kg << 6) + (ls << 5);
        a.q[0] = *(const u32x4*)ab;          // K 16*ls + 0..7   (contiguous, matches B)
        a.q[1] = *(const u32x4*)(ab + 16);   // K 16*ls + 8..15
        #pragma unroll
        for (int g = 0; g < 4; ++g)
          C[g][m] = __builtin_amdgcn_wmma_f32_16x16x32_bf16(
              false, a.v, false, Bg[g], (short)0, C[g][m], false, false);
      }
    }
    __syncthreads();                    // all A reads done before h is rewritten

    // ---- LSTM cell update (i,f,g,o at identical lane/VGPR slots across C[0..3]) ----
    #pragma unroll
    for (int m = 0; m < 4; ++m) {
      #pragma unroll
      for (int r = 0; r < 8; ++r) {
        int row = m * 16 + r + 8 * ls;
        if (t < slenL[row]) {
          float iv = C[0][m][r], fv = C[1][m][r], gv = C[2][m][r], ov = C[3][m][r];
          float cold = csL[row * H_DIM + jcol];
          float cn = sigmf(fv) * cold + sigmf(iv) * tanh_fast(gv);
          float hn = sigmf(ov) * tanh_fast(cn);
          csL[row * H_DIM + jcol] = cn;
          *(unsigned short*)(Abuf + row * A_STRIDE + 512 + jcol * 2) = f2bf(hn); // bf16 h for next matmul
          h32[row * H_DIM + jcol] = hn;                                          // f32 h for output
        }
      }
    }
    // stagers write x region, updaters write h/cs regions (disjoint); next barrier orders both
  }
  __syncthreads();

  // ---- write final h (f32) ----
  #pragma unroll
  for (int i = 0; i < 32; ++i) {
    int idx = tid + i * NTHREADS;       // 16384 elements
    int row = idx >> 8, col = idx & 255;
    out[((size_t)(row0 + row)) * H_DIM + col] = h32[idx];
  }
  // ---- last timestamps (as float; exact for ts < 2^24) ----
  if (tid < M_TILE) {
    int n   = row0 + tid;
    int len = slenL[tid];
    if (len < 1) len = 1;
    out[(size_t)N_SEQ * H_DIM + n] = (float)tstamps[n * T_SEQ + (len - 1)];
  }
}

// ---------------- launch ----------------
extern "C" void kernel_launch(void* const* d_in, const int* in_sizes, int n_in,
                              void* d_out, int out_size, void* d_ws, size_t ws_size,
                              hipStream_t stream) {
  (void)in_sizes; (void)n_in; (void)out_size; (void)ws_size;
  const float* msg      = (const float*)d_in[0];
  const float* Wih      = (const float*)d_in[1];
  const float* Whh      = (const float*)d_in[2];
  const float* bih      = (const float*)d_in[3];
  const float* bhh      = (const float*)d_in[4];
  const int*   lengths  = (const int*)d_in[5];
  const int*   tstamps  = (const int*)d_in[6];
  float*       out      = (float*)d_out;

  unsigned short* wpk  = (unsigned short*)d_ws;                                  // 1 MB bf16 weights
  float*          bias = (float*)((char*)d_ws + (size_t)G_DIM * K_TOT * sizeof(unsigned short));

  lstm_pack_weights<<<(G_DIM * K_TOT + 255) / 256, 256, 0, stream>>>(Wih, Whh, bih, bhh, wpk, bias);

  size_t shmem = LDS_SIZE;                                                       // ~193 KB
  hipFuncSetAttribute((const void*)lstm_wmma_kernel,
                      hipFuncAttributeMaxDynamicSharedMemorySize, (int)shmem);
  lstm_wmma_kernel<<<N_SEQ / M_TILE, NTHREADS, shmem, stream>>>(
      msg, lengths, tstamps, wpk, bias, out);
}
